// RelativeAttention_48026324303907
// MI455X (gfx1250) — matrix operand order repaired and checked by reading, not compiled, not run on hardware
//
#include <hip/hip_runtime.h>
#include <stdint.h>

#define SEQL  2048
#define DM    1024
#define NH    16
#define DH    64
#define BATCH 2
#define MROWS (BATCH*SEQL)   // 4096

typedef __bf16 bf16x16 __attribute__((ext_vector_type(16)));
typedef float  f32x8   __attribute__((ext_vector_type(8)));

union BFrag { bf16x16 v; unsigned int u[8]; };

__device__ __forceinline__ unsigned short f2bf(float f) {
    unsigned int u = __float_as_uint(f);
    u = (u + 0x7FFFu + ((u >> 16) & 1u)) >> 16;
    return (unsigned short)u;
}
__device__ __forceinline__ unsigned int pack2bf(float lo, float hi) {
    return (unsigned int)f2bf(lo) | ((unsigned int)f2bf(hi) << 16);
}
__device__ __forceinline__ f32x8 wmma_bf16(const BFrag& a, const BFrag& b, f32x8 c) {
    return __builtin_amdgcn_wmma_f32_16x16x32_bf16(false, a.v, false, b.v,
                                                   (short)0, c, false, false);
}

// CDNA5 async global->LDS copy (ASYNCcnt tracked, bypasses VGPRs). GV mode.
__device__ __forceinline__ void async_ld_b128(unsigned ldsOff, const void* g) {
    asm volatile("global_load_async_to_lds_b128 %0, %1, off"
                 :: "v"(ldsOff), "v"((unsigned long long)(uintptr_t)g)
                 : "memory");
}
#define WAIT_ASYNC(N) asm volatile("s_wait_asynccnt %0" :: "i"(N) : "memory")

// ---------------- conversion kernels ----------------
__global__ void cvt_bf16_kernel(const float* __restrict__ in,
                                unsigned short* __restrict__ out, int n) {
    int i = blockIdx.x * blockDim.x + threadIdx.x;
    if (i < n) out[i] = f2bf(in[i]);
}

// in[rows][cols] (f32) -> out[cols][rows] (bf16)
__global__ void cvt_transpose_kernel(const float* __restrict__ in,
                                     unsigned short* __restrict__ out,
                                     int rows, int cols) {
    int i = blockIdx.x * blockDim.x + threadIdx.x;
    if (i < rows * cols) {
        int r = i / cols, c = i % cols;
        out[(size_t)c * rows + r] = f2bf(in[i]);
    }
}

// ---------------- GEMM: C[m,n] = sum_k A[m,k] * Wt[k,n] + bias[n] ----------------
// A: [4096][1024] bf16.  Wt: [1024][1024] bf16 (pre-transposed weight).
// 128x128 block tile, K=32 steps, double-buffered LDS fed by async loads.
// mode 0: bf16 head-major [B][H][S][DH]  (Q, V)
// mode 2: bf16 d-major    [B][H][DH][S]  (K, pre-transposed for attention)
// mode 1: f32 row-major + bias           (final projection)
__global__ __launch_bounds__(256) void gemm_bf16_kernel(
    const unsigned short* __restrict__ A,
    const unsigned short* __restrict__ Wt,
    const float* __restrict__ bias,
    unsigned short* __restrict__ outBF,
    float* __restrict__ outF32,
    int mode) {
    __shared__ unsigned short As[2][128 * 32];
    __shared__ unsigned short Bs[2][32 * 128];

    const int tid  = threadIdx.x;
    const int lane = tid & 31;
    const int wv   = tid >> 5;
    const int m0   = blockIdx.x * 128;
    const int n0   = blockIdx.y * 128;
    const int ml   = lane & 15;
    const int half = lane >> 4;

    f32x8 acc[8];
#pragma unroll
    for (int i = 0; i < 8; ++i)
#pragma unroll
        for (int r = 0; r < 8; ++r) acc[i][r] = 0.0f;

    const int aRow = tid >> 1;         // 0..127
    const int aCol = (tid & 1) * 16;   // 0 or 16
    const int bRow = tid >> 3;         // 0..31
    const int bCol = (tid & 7) * 16;   // 0..112

    const unsigned short* aG = A  + (size_t)(m0 + aRow) * DM + aCol;
    const unsigned short* bG = Wt + (size_t)bRow        * DM + n0 + bCol;

    auto stage_loads = [&](int kt, int stage) {
        const int k0 = kt * 32;
        unsigned asOff = (unsigned)(uintptr_t)(&As[stage][aRow * 32  + aCol]);
        unsigned bsOff = (unsigned)(uintptr_t)(&Bs[stage][bRow * 128 + bCol]);
        async_ld_b128(asOff,      aG + k0);
        async_ld_b128(asOff + 16, aG + k0 + 8);
        async_ld_b128(bsOff,      bG + (size_t)k0 * DM);
        async_ld_b128(bsOff + 16, bG + (size_t)k0 * DM + 8);
    };

    stage_loads(0, 0);
    for (int kt = 0; kt < 32; ++kt) {
        const int cur = kt & 1;
        if (kt + 1 < 32) {
            stage_loads(kt + 1, cur ^ 1);
            WAIT_ASYNC(4);             // current stage complete, next in flight
        } else {
            WAIT_ASYNC(0);
        }
        __syncthreads();

        BFrag af;
#pragma unroll
        for (int v = 0; v < 8; ++v) {
            int kk = ((v >> 2) << 4) + half * 8 + ((v & 3) << 1);
            af.u[v] = *(const unsigned int*)(&As[cur][(wv * 16 + ml) * 32 + kk]);
        }
#pragma unroll
        for (int nc = 0; nc < 8; ++nc) {
            BFrag bf;
#pragma unroll
            for (int v = 0; v < 8; ++v) {
                int kk = ((v >> 2) << 4) + half * 8 + ((v & 3) << 1);
                bf.u[v] = (unsigned int)Bs[cur][kk * 128 + nc * 16 + ml] |
                          ((unsigned int)Bs[cur][(kk + 1) * 128 + nc * 16 + ml] << 16);
            }
            acc[nc] = wmma_bf16(af, bf, acc[nc]);
        }
        __syncthreads();
    }

#pragma unroll
    for (int nc = 0; nc < 8; ++nc) {
        int gcol = n0 + nc * 16 + ml;
        float bv = bias[gcol];
#pragma unroll
        for (int r = 0; r < 8; ++r) {
            int grow = m0 + wv * 16 + r + 8 * half;
            float val = acc[nc][r] + bv;
            int b = grow >> 11;        // / SEQL
            int s = grow & (SEQL - 1);
            int h = gcol >> 6;
            int d = gcol & 63;
            if (mode == 0) {
                outBF[(((size_t)(b * NH + h)) * SEQL + s) * DH + d] = f2bf(val);
            } else if (mode == 2) {
                outBF[(((size_t)(b * NH + h)) * DH + d) * SEQL + s] = f2bf(val);
            } else {
                outF32[(size_t)grow * DM + gcol] = val;
            }
        }
    }
}

// ---------------- flash attention with fused relative-position skew ----------------
#define AT_WAVES 4
__global__ __launch_bounds__(128) void attn_kernel(
    const unsigned short* __restrict__ Qb,   // [BH][S][DH] bf16
    const unsigned short* __restrict__ Kt,   // [BH][DH][S] bf16 (d-major)
    const unsigned short* __restrict__ Vb,   // [BH][S][DH] bf16
    const unsigned short* __restrict__ Ert,  // [DH][SEQL] bf16 (Er transposed)
    unsigned short* __restrict__ Ob) {       // [B][S][DM] bf16
    __shared__ float Pl[AT_WAVES][16 * 32];
    __shared__ float Rl[AT_WAVES][16 * 48];

    const int lane = threadIdx.x & 31;
    const int w    = threadIdx.x >> 5;
    const int tile = blockIdx.x * AT_WAVES + w;
    const int bh   = tile >> 7;          // / (S/16)
    const int st   = tile & 127;
    const int s0   = st * 16;
    const int ml   = lane & 15;
    const int half = lane >> 4;
    const float scale = 0.125f;          // 1/sqrt(DH)

    // Q fragments (A layout, direct from global; rows contiguous in d)
    BFrag qa0, qa1;
    const unsigned short* qrow = Qb + ((size_t)bh * SEQL + s0 + ml) * DH;
#pragma unroll
    for (int v = 0; v < 8; ++v) {
        int kk = ((v >> 2) << 4) + half * 8 + ((v & 3) << 1);
        qa0.u[v] = *(const unsigned int*)(qrow + kk);
        qa1.u[v] = *(const unsigned int*)(qrow + 32 + kk);
    }

    // K base rows (B operand: lane = n = t, elements walk d = k in A's order)
    const unsigned short* krow0 = Kt + ((size_t)bh * DH)      * SEQL;
    const unsigned short* krow1 = Kt + ((size_t)bh * DH + 32) * SEQL;
    const unsigned short* erow0 = Ert;
    const unsigned short* erow1 = Ert + (size_t)32 * SEQL;

    float rmax[8], rsum[8], scarr[8];
    f32x8 accO[4];
#pragma unroll
    for (int r = 0; r < 8; ++r) { rmax[r] = -3.0e38f; rsum[r] = 0.0f; }
#pragma unroll
    for (int nc = 0; nc < 4; ++nc)
#pragma unroll
        for (int r = 0; r < 8; ++r) accO[nc][r] = 0.0f;

    const int nkb = ((s0 + 15) >> 5) + 1;
    for (int kb = 0; kb < nkb; ++kb) {
        const int t0 = kb * 32;

        // Q @ K^T (16x32 logits) — K loaded directly in B layout from d-major K
        f32x8 accS[2];
#pragma unroll
        for (int nc = 0; nc < 2; ++nc) {
#pragma unroll
            for (int r = 0; r < 8; ++r) accS[nc][r] = 0.0f;
            BFrag b0, b1;
#pragma unroll
            for (int v = 0; v < 8; ++v) {
                int kk = ((v >> 2) << 4) + half * 8 + ((v & 3) << 1);
                int tc = t0 + nc * 16 + ml;
                b0.u[v] = (unsigned int)krow0[(size_t)kk * SEQL + tc] | ((unsigned int)krow0[(size_t)(kk + 1) * SEQL + tc] << 16);
                b1.u[v] = (unsigned int)krow1[(size_t)kk * SEQL + tc] | ((unsigned int)krow1[(size_t)(kk + 1) * SEQL + tc] << 16);
            }
            accS[nc] = wmma_bf16(qa0, b0, accS[nc]);
            accS[nc] = wmma_bf16(qa1, b1, accS[nc]);
        }

        // relative term: R[m,u] = q[m] . Er[base+u], u in [0,48)
        const int base = (SEQL - 16) - s0 + t0;
        f32x8 accR[3];
#pragma unroll
        for (int uc = 0; uc < 3; ++uc) {
#pragma unroll
            for (int r = 0; r < 8; ++r) accR[uc][r] = 0.0f;
            BFrag e0, e1;
#pragma unroll
            for (int v = 0; v < 8; ++v) {
                int kk = ((v >> 2) << 4) + half * 8 + ((v & 3) << 1);
                int col = base + uc * 16 + ml;
                if (col > SEQL - 1) col = SEQL - 1;  // OOB only in masked region
                e0.u[v] = (unsigned int)erow0[(size_t)kk * SEQL + col] | ((unsigned int)erow0[(size_t)(kk + 1) * SEQL + col] << 16);
                e1.u[v] = (unsigned int)erow1[(size_t)kk * SEQL + col] | ((unsigned int)erow1[(size_t)(kk + 1) * SEQL + col] << 16);
            }
            accR[uc] = wmma_bf16(qa0, e0, accR[uc]);
            accR[uc] = wmma_bf16(qa1, e1, accR[uc]);
        }
        // spill R to LDS for the skewed gather
#pragma unroll
        for (int uc = 0; uc < 3; ++uc)
#pragma unroll
            for (int r = 0; r < 8; ++r)
                Rl[w][(r + 8 * half) * 48 + uc * 16 + ml] = accR[uc][r];

        // logits + causal mask + online softmax
        float lg0[8], lg1[8];
#pragma unroll
        for (int r = 0; r < 8; ++r) {
            int m = r + 8 * half;
            int s = s0 + m;
            int t = t0 + ml;
            float l0 = (accS[0][r] + Rl[w][m * 48 + 15 + ml - m]) * scale;
            lg0[r] = (t <= s) ? l0 : -1.0e9f;
            t = t0 + 16 + ml;
            float l1 = (accS[1][r] + Rl[w][m * 48 + 31 + ml - m]) * scale;
            lg1[r] = (t <= s) ? l1 : -1.0e9f;
        }
#pragma unroll
        for (int r = 0; r < 8; ++r) {
            float c = fmaxf(lg0[r], lg1[r]);
#pragma unroll
            for (int ofs = 1; ofs < 16; ofs <<= 1)
                c = fmaxf(c, __shfl_xor(c, ofs, 32));
            float nm = fmaxf(rmax[r], c);
            float sc = __expf(rmax[r] - nm);
            float p0 = __expf(lg0[r] - nm);
            float p1 = __expf(lg1[r] - nm);
            float ps = p0 + p1;
#pragma unroll
            for (int ofs = 1; ofs < 16; ofs <<= 1)
                ps += __shfl_xor(ps, ofs, 32);
            rsum[r] = rsum[r] * sc + ps;
            rmax[r] = nm;
            scarr[r] = sc;
            int m = r + 8 * half;
            Pl[w][m * 32 + ml]      = p0;
            Pl[w][m * 32 + 16 + ml] = p1;
        }
#pragma unroll
        for (int nc = 0; nc < 4; ++nc)
#pragma unroll
            for (int r = 0; r < 8; ++r) accO[nc][r] *= scarr[r];

        // P (C layout f32) -> A layout bf16 via wave-private LDS
        BFrag pf;
#pragma unroll
        for (int v = 0; v < 8; ++v) {
            int kk = ((v >> 2) << 4) + half * 8 + ((v & 3) << 1);
            pf.u[v] = pack2bf(Pl[w][ml * 32 + kk], Pl[w][ml * 32 + kk + 1]);
        }
        // P @ V  (V as B operand direct from global: lane = d, elements walk t_local)
#pragma unroll
        for (int nc = 0; nc < 4; ++nc) {
            BFrag vf;
#pragma unroll
            for (int v = 0; v < 8; ++v) {
                int kk = ((v >> 2) << 4) + half * 8 + ((v & 3) << 1);
                const unsigned short* vp = Vb + ((size_t)bh * SEQL + t0 + kk) * DH + nc * 16 + ml;
                vf.u[v] = (unsigned int)vp[0] | ((unsigned int)vp[DH] << 16);
            }
            accO[nc] = wmma_bf16(pf, vf, accO[nc]);
        }
    }

    // write O in [B][S][DM] bf16 (row-major for output projection)
    const int b = bh >> 4;
    const int h = bh & 15;
#pragma unroll
    for (int nc = 0; nc < 4; ++nc) {
        int d = nc * 16 + ml;
#pragma unroll
        for (int r = 0; r < 8; ++r) {
            int m = r + 8 * half;
            float val = accO[nc][r] / rsum[r];
            Ob[((size_t)(b * SEQL + s0 + m)) * DM + h * DH + d] = f2bf(val);
        }
    }
}

// ---------------- host launcher ----------------
extern "C" void kernel_launch(void* const* d_in, const int* in_sizes, int n_in,
                              void* d_out, int out_size, void* d_ws, size_t ws_size,
                              hipStream_t stream) {
    (void)in_sizes; (void)n_in; (void)out_size; (void)ws_size;
    const float* x  = (const float*)d_in[0];
    const float* Wq = (const float*)d_in[1];
    const float* bq = (const float*)d_in[2];
    const float* Wk = (const float*)d_in[3];
    const float* bk = (const float*)d_in[4];
    const float* Wv = (const float*)d_in[5];
    const float* bv = (const float*)d_in[6];
    const float* Er = (const float*)d_in[7];
    const float* Wo = (const float*)d_in[8];
    const float* bo = (const float*)d_in[9];
    float* out = (float*)d_out;

    char* wp = (char*)d_ws;
    auto alloc = [&](size_t bytes) -> unsigned short* {
        unsigned short* p = (unsigned short*)wp;
        wp += (bytes + 255) & ~(size_t)255;
        return p;
    };
    unsigned short* Xb  = alloc((size_t)MROWS * DM * 2);
    unsigned short* Wqt = alloc((size_t)DM * DM * 2);
    unsigned short* Wkt = alloc((size_t)DM * DM * 2);
    unsigned short* Wvt = alloc((size_t)DM * DM * 2);
    unsigned short* Wot = alloc((size_t)DM * DM * 2);
    unsigned short* Ert = alloc((size_t)DH * SEQL * 2);
    unsigned short* Qb  = alloc((size_t)MROWS * DM * 2);
    unsigned short* Ktw = alloc((size_t)MROWS * DM * 2);
    unsigned short* Vb  = alloc((size_t)MROWS * DM * 2);
    unsigned short* Ob  = alloc((size_t)MROWS * DM * 2);

    cvt_bf16_kernel<<<(MROWS * DM) / 256, 256, 0, stream>>>(x, Xb, MROWS * DM);
    cvt_transpose_kernel<<<(DM * DM) / 256, 256, 0, stream>>>(Wq, Wqt, DM, DM);
    cvt_transpose_kernel<<<(DM * DM) / 256, 256, 0, stream>>>(Wk, Wkt, DM, DM);
    cvt_transpose_kernel<<<(DM * DM) / 256, 256, 0, stream>>>(Wv, Wvt, DM, DM);
    cvt_transpose_kernel<<<(DM * DM) / 256, 256, 0, stream>>>(Wo, Wot, DM, DM);
    cvt_transpose_kernel<<<(SEQL * DH) / 256, 256, 0, stream>>>(Er, Ert, SEQL, DH);

    dim3 g(MROWS / 128, DM / 128), blk(256);
    gemm_bf16_kernel<<<g, blk, 0, stream>>>(Xb, Wqt, bq, Qb,  nullptr, 0);
    gemm_bf16_kernel<<<g, blk, 0, stream>>>(Xb, Wkt, bk, Ktw, nullptr, 2);
    gemm_bf16_kernel<<<g, blk, 0, stream>>>(Xb, Wvt, bv, Vb,  nullptr, 0);

    attn_kernel<<<(BATCH * NH * (SEQL / 16)) / AT_WAVES, 32 * AT_WAVES, 0, stream>>>(
        Qb, Ktw, Vb, Ert, Ob);

    gemm_bf16_kernel<<<g, blk, 0, stream>>>(Ob, Wot, bo, nullptr, out, 1);
}
